// DSGNet_50448685859251
// MI455X (gfx1250) — matrix operand order repaired and checked by reading, not compiled, not run on hardware
//
#include <hip/hip_runtime.h>
#include <hip/hip_bf16.h>
#include <math.h>

// ---------------- types / helpers ----------------
typedef __attribute__((ext_vector_type(16))) __bf16 v16bf;
typedef __attribute__((ext_vector_type(8)))  float  v8f;

__device__ __forceinline__ unsigned short f32_to_bf16_rne(float f) {
  unsigned u = __float_as_uint(f);
  unsigned r = u + 0x7FFFu + ((u >> 16) & 1u);
  return (unsigned short)(r >> 16);
}

// ---------------- elementwise converts ----------------
__global__ void k_cvt_bf16(const float* __restrict__ in, unsigned short* __restrict__ out, int n) {
  int i = blockIdx.x * 256 + threadIdx.x;
  if (i < n) out[i] = f32_to_bf16_rne(in[i]);
}

// in: [R][C] f32 row-major  ->  out: [C][R] bf16 row-major (pre-transposed B operand)
__global__ void k_cvtT_bf16(const float* __restrict__ in, unsigned short* __restrict__ out, int R, int C) {
  int i = blockIdx.x * 256 + threadIdx.x;
  if (i < R * C) {
    int c = i / R;
    int r = i - c * R;
    out[i] = f32_to_bf16_rne(in[(size_t)r * C + c]);
  }
}

__global__ void k_copy_f32(const float* __restrict__ in, float* __restrict__ out, int n) {
  int i = blockIdx.x * 256 + threadIdx.x;
  if (i < n) out[i] = in[i];
}

__global__ void k_zero_f32(float* p, int n) {
  int i = blockIdx.x * 256 + threadIdx.x;
  if (i < n) p[i] = 0.f;
}

// ---------------- WMMA GEMM: C[M,N] = act(A[M,K] @ B^T[N,K] + bias) ----------------
// A, B: bf16 row-major (B is stored transposed: [N][K]). One wave per 16x16 tile.
// Fragment layouts per CDNA5 ISA 7.12.2 (wave32):
//   A: lane<16 -> row=lane, elems {k0..k0+7, k0+16..k0+23}; lane>=16 -> row=lane-16, {k0+8..15, k0+24..31}
//   B: lane<16 -> col=lane, elems {k0..k0+7, k0+16..k0+23}; lane>=16 -> col=lane-16, {k0+8..15, k0+24..31}
//   C: VGPR j -> row = (lane>=16 ? j+8 : j), col = lane&15
template <int ACT /*0=none,1=tanh,2=relu*/, bool HAS_BIAS, bool ST_F32, bool ST_BF16>
__global__ __launch_bounds__(256) void k_gemm_bt(
    const unsigned short* __restrict__ A, const unsigned short* __restrict__ B,
    const float* __restrict__ bias, float* __restrict__ C, unsigned short* __restrict__ Cbf,
    int M, int N, int Kd) {
  int lane = threadIdx.x & 31;
  int wave = threadIdx.x >> 5;
  long tile = (long)blockIdx.x * 8 + wave;
  int tilesN = N >> 4;
  long nTiles = (long)(M >> 4) * tilesN;
  if (tile >= nTiles) return;
  int tm = (int)(tile / tilesN);
  int tn = (int)(tile % tilesN);

  int mr    = lane & 15;
  int koffA = (lane >> 4) * 8;
  int koffB = (lane >> 4) * 8;
  const unsigned short* ap = A + (size_t)(tm * 16 + mr) * Kd + koffA;
  const unsigned short* bp = B + (size_t)(tn * 16 + mr) * Kd + koffB;

  v8f acc = {0.f, 0.f, 0.f, 0.f, 0.f, 0.f, 0.f, 0.f};
  for (int k0 = 0; k0 < Kd; k0 += 32) {
    union { uint4 u[2]; v16bf v; } fa, fb;
    fa.u[0] = *(const uint4*)(ap + k0);
    fa.u[1] = *(const uint4*)(ap + k0 + 16);
    fb.u[0] = *(const uint4*)(bp + k0);
    fb.u[1] = *(const uint4*)(bp + k0 + 16);
    __builtin_prefetch((const void*)(ap + k0 + 128), 0, 1);
    __builtin_prefetch((const void*)(bp + k0 + 128), 0, 1);
    acc = __builtin_amdgcn_wmma_f32_16x16x32_bf16(false, fa.v, false, fb.v,
                                                  (short)0, acc, false, false);
  }

  int cn = tn * 16 + (lane & 15);
  int rb = tm * 16 + (lane >> 4) * 8;
  float bv = HAS_BIAS ? bias[cn] : 0.f;
#pragma unroll
  for (int j = 0; j < 8; ++j) {
    float v = acc[j] + bv;
    if (ACT == 1) v = tanhf(v);
    if (ACT == 2) v = fmaxf(v, 0.f);
    size_t off = (size_t)(rb + j) * N + cn;
    if (ST_F32) C[off] = v;
    if (ST_BF16) Cbf[off] = f32_to_bf16_rne(v);
  }
}

// ---------------- dual GEMV: o1[r] = X[r,:]@v1 ; o2[r] = X[r,:]@v2  (H=256) ----------------
__global__ __launch_bounds__(256) void k_gemv2(const float* __restrict__ X,
                                               const float* __restrict__ v1,
                                               const float* __restrict__ v2,
                                               float* __restrict__ o1, float* __restrict__ o2, int M) {
  int lane = threadIdx.x & 31;
  int wave = threadIdx.x >> 5;
  int r = blockIdx.x * 8 + wave;
  if (r >= M) return;
  const float* x = X + (size_t)r * 256;
  float s1 = 0.f, s2 = 0.f;
#pragma unroll
  for (int t = 0; t < 8; ++t) {
    int c = t * 32 + lane;
    float xv = x[c];
    s1 += xv * v1[c];
    s2 += xv * v2[c];
  }
#pragma unroll
  for (int off = 16; off; off >>= 1) {
    s1 += __shfl_xor(s1, off, 32);
    s2 += __shfl_xor(s2, off, 32);
  }
  if (lane == 0) { o1[r] = s1; o2[r] = s2; }
}

// ---------------- per-entity edge score + top8 + softmax + weighted compose ----------------
// One wave per entity: lane k handles edge n*32+k. Output neigh (bf16, [N_ENT,256]).
__global__ __launch_bounds__(256) void k_edge_topk(
    const int* __restrict__ esrc, const int* __restrict__ erel,
    const float* __restrict__ ha1, const float* __restrict__ ha2, const float* __restrict__ ha3,
    const float* __restrict__ h, const float* __restrict__ hr,
    unsigned short* __restrict__ neigh_bf, int nEnt) {
  int lane = threadIdx.x & 31;
  int wave = threadIdx.x >> 5;
  int n = blockIdx.x * 8 + wave;
  if (n >= nEnt) return;
  long e = (long)n * 32 + lane;
  int se = esrc[e];
  int re = erel[e];
  float s = ha1[se] + ha2[n] + ha3[re];
  s = (s > 0.f) ? s : 0.2f * s;  // leaky relu

  float v = s;
  float topv[8];
  int topsrc[8], toprel[8];
#pragma unroll
  for (int t = 0; t < 8; ++t) {  // argmax with tie -> lowest index (matches top_k order)
    float m = v;
    int mi = lane;
#pragma unroll
    for (int off = 16; off; off >>= 1) {
      float om = __shfl_xor(m, off, 32);
      int omi = __shfl_xor(mi, off, 32);
      if (om > m || (om == m && omi < mi)) { m = om; mi = omi; }
    }
    topv[t] = m;
    topsrc[t] = __shfl(se, mi, 32);
    toprel[t] = __shfl(re, mi, 32);
    if (lane == mi) v = -3.4e38f;
  }

  float vmax = topv[0];
  float sum = 0.f;
  float al[8];
#pragma unroll
  for (int t = 0; t < 8; ++t) { al[t] = __expf(topv[t] - vmax); sum += al[t]; }
  float inv = 1.f / sum;

  float acc[8] = {0.f, 0.f, 0.f, 0.f, 0.f, 0.f, 0.f, 0.f};
#pragma unroll
  for (int t = 0; t < 8; ++t) {
    const float* hp = h + (size_t)topsrc[t] * 256;
    const float* rp = hr + (size_t)toprel[t] * 256;
    float aw = al[t] * inv;
#pragma unroll
    for (int j = 0; j < 8; ++j) {
      int c = j * 32 + lane;
      acc[j] += aw * hp[c] * rp[c];
    }
  }
#pragma unroll
  for (int j = 0; j < 8; ++j)
    neigh_bf[(size_t)n * 256 + j * 32 + lane] = f32_to_bf16_rne(acc[j]);
}

// ---------------- correlation statistics: colsums + global sum-of-products ----------------
// stats layout: [0..255]=colsum(cc) [256..511]=colsum(cp) [512]=S11 [513]=S22 [514]=S12
__global__ __launch_bounds__(256) void k_stats(const float* __restrict__ cc, const float* __restrict__ cp,
                                               float* __restrict__ stats, int nEnt) {
  int j = threadIdx.x;
  float c1 = 0.f, c2 = 0.f, s11 = 0.f, s22 = 0.f, s12 = 0.f;
  for (int r = blockIdx.x; r < nEnt; r += gridDim.x) {
    float a = cc[(size_t)r * 256 + j];
    float b = cp[(size_t)r * 256 + j];
    c1 += a; c2 += b;
    s11 += a * a; s22 += b * b; s12 += a * b;
  }
  atomicAdd(&stats[j], c1);
  atomicAdd(&stats[256 + j], c2);
  __shared__ float sd[256];
  sd[j] = s11; __syncthreads();
  for (int st = 128; st; st >>= 1) { if (j < st) sd[j] += sd[j + st]; __syncthreads(); }
  if (j == 0) atomicAdd(&stats[512], sd[0]);
  __syncthreads();
  sd[j] = s22; __syncthreads();
  for (int st = 128; st; st >>= 1) { if (j < st) sd[j] += sd[j + st]; __syncthreads(); }
  if (j == 0) atomicAdd(&stats[513], sd[0]);
  __syncthreads();
  sd[j] = s12; __syncthreads();
  for (int st = 128; st; st >>= 1) { if (j < st) sd[j] += sd[j + st]; __syncthreads(); }
  if (j == 0) atomicAdd(&stats[514], sd[0]);
}

__global__ void k_finalize(const float* __restrict__ st0, const float* __restrict__ st1,
                           float* __restrict__ out_corr, int nEnt) {
  float total = 0.f;
  for (int i = 0; i < 2; ++i) {
    const float* s = i ? st1 : st0;
    float mm12 = 0.f, mm11 = 0.f, mm22 = 0.f;
    for (int j = 0; j < 256; ++j) {
      float m1 = s[j] / nEnt;
      float m2 = s[256 + j] / nEnt;
      mm12 += m1 * m2; mm11 += m1 * m1; mm22 += m2 * m2;
    }
    float NH = (float)nEnt * 256.f;
    float cov  = (s[514] - nEnt * mm12) / NH;
    float var1 = (s[512] - nEnt * mm11) / NH;
    float var2 = (s[513] - nEnt * mm22) / NH;
    float sd1 = sqrtf(fmaxf(var1, 0.f));
    float sd2 = sqrtf(fmaxf(var2, 0.f));
    total += fabsf(cov) / (sd1 * sd2 + 1e-8f);
  }
  out_corr[0] = total * 0.5f;
}

// ---------------- fusion accumulators (softmax(gate) computed in-kernel) ----------------
__global__ void k_accum(const float* __restrict__ cc, const float* __restrict__ cp,
                        const float* __restrict__ gate, int gi, float* __restrict__ ef, int n) {
  int i = blockIdx.x * 256 + threadIdx.x;
  if (i >= n) return;
  float g0 = gate[0], g1 = gate[1];
  float m = fmaxf(g0, g1);
  float e0 = __expf(g0 - m), e1 = __expf(g1 - m);
  float g = (gi ? e1 : e0) / (e0 + e1);
  ef[i] += g * (cc[i] + cp[i]);
}

__global__ void k_relfused(const float* __restrict__ rel, const float* __restrict__ gate,
                           float* __restrict__ rf, int n) {
  int i = blockIdx.x * 256 + threadIdx.x;
  if (i >= n) return;
  float g0 = gate[0], g1 = gate[1];
  float m = fmaxf(g0, g1);
  float e0 = __expf(g0 - m), e1 = __expf(g1 - m);
  float inv = 1.f / (e0 + e1);
  rf[i] = (e0 * inv) * rel[i] + (e1 * inv) * rel[n + i];
}

// ---------------- 7x7 VALID conv over stacked 32x16 image, relu, store bf16 ----------------
__global__ __launch_bounds__(256) void k_conv(const int* __restrict__ h_id, const int* __restrict__ r_id,
                                              const float* __restrict__ ef, const float* __restrict__ rf,
                                              const float* __restrict__ cw, const float* __restrict__ cb,
                                              unsigned short* __restrict__ xbf) {
  __shared__ float img[512];
  __shared__ float w[32 * 49];
  __shared__ float bsh[32];
  int b = blockIdx.x;
  int tid = threadIdx.x;
  int hid = h_id[b];
  int rid = r_id[b];
  img[tid]       = ef[(size_t)hid * 256 + tid];
  img[256 + tid] = rf[(size_t)rid * 256 + tid];
  for (int i = tid; i < 32 * 49; i += 256) w[i] = cw[i];
  if (tid < 32) bsh[tid] = cb[tid];
  __syncthreads();
  for (int o = tid; o < 8320; o += 256) {
    int oc = o / 260;
    int rem = o - oc * 260;
    int oy = rem / 10;
    int ox = rem - oy * 10;
    float s = bsh[oc];
    const float* wk = &w[oc * 49];
#pragma unroll
    for (int ky = 0; ky < 7; ++ky)
#pragma unroll
      for (int kx = 0; kx < 7; ++kx)
        s += img[(oy + ky) * 16 + ox + kx] * wk[ky * 7 + kx];
    s = fmaxf(s, 0.f);
    xbf[(size_t)b * 8320 + o] = f32_to_bf16_rne(s);
  }
}

// ---------------- host orchestration ----------------
extern "C" void kernel_launch(void* const* d_in, const int* in_sizes, int n_in,
                              void* d_out, int out_size, void* d_ws, size_t ws_size,
                              hipStream_t stream) {
  (void)in_sizes; (void)n_in; (void)out_size; (void)ws_size;
  const int*   h_id = (const int*)d_in[0];
  const int*   r_id = (const int*)d_in[1];
  const int*   esrc = (const int*)d_in[2];
  const int*   erel = (const int*)d_in[3];
  const float* ent  = (const float*)d_in[4];
  const float* rel  = (const float*)d_in[5];
  const float* gate = (const float*)d_in[6];
  const float* S_w  = (const float*)d_in[7];
  const float* S_b  = (const float*)d_in[8];
  const float* L_w  = (const float*)d_in[9];
  const float* L_b  = (const float*)d_in[10];
  const float* W    = (const float*)d_in[11];
  const float* W_r  = (const float*)d_in[12];
  const float* a    = (const float*)d_in[13];
  const float* nw   = (const float*)d_in[14];
  const float* cw   = (const float*)d_in[15];
  const float* cb   = (const float*)d_in[16];
  const float* fcw  = (const float*)d_in[17];
  const float* fcb  = (const float*)d_in[18];
  const float* ebias= (const float*)d_in[19];

  const int NE = 20000, H = 256;
  const int EH = NE * H;          // 5,120,000
  const int RELN = 2 * 200 * H;   // 102,400 per layer slice
  const int FLAT = 8320, BS = 512;

  char* base = (char*)d_ws;
  size_t off = 0;
  auto alloc = [&](size_t bytes) -> char* {
    char* p = base + off;
    off = (off + bytes + 255) & ~(size_t)255;
    return p;
  };

  unsigned short* entbf  = (unsigned short*)alloc((size_t)EH * 2);
  unsigned short* relbf  = (unsigned short*)alloc((size_t)2 * RELN * 2);
  unsigned short* swT    = (unsigned short*)alloc(65536 * 2);
  unsigned short* lwT    = (unsigned short*)alloc(65536 * 2);
  unsigned short* wT     = (unsigned short*)alloc((size_t)2 * 65536 * 2);
  unsigned short* wrT    = (unsigned short*)alloc((size_t)2 * 65536 * 2);
  unsigned short* nwT    = (unsigned short*)alloc((size_t)2 * 65536 * 2);
  unsigned short* fcT    = (unsigned short*)alloc((size_t)256 * FLAT * 2);
  float*          common = (float*)alloc((size_t)EH * 4);
  unsigned short* commbf = (unsigned short*)alloc((size_t)EH * 2);
  float*          priv   = (float*)alloc((size_t)EH * 4);
  unsigned short* privbf = (unsigned short*)alloc((size_t)EH * 2);
  float*          hbuf   = (float*)alloc((size_t)EH * 4);
  float*          hrbuf  = (float*)alloc((size_t)RELN * 4);
  float*          ha1    = (float*)alloc((size_t)NE * 4);
  float*          ha2    = (float*)alloc((size_t)NE * 4);
  float*          ha3    = (float*)alloc(400 * 4);
  float*          dummy  = (float*)alloc(400 * 4);
  unsigned short* neighbf= (unsigned short*)alloc((size_t)EH * 2);
  float*          ccb    = (float*)alloc((size_t)EH * 4);
  float*          cpb    = (float*)alloc((size_t)EH * 4);
  float*          efused = (float*)alloc((size_t)EH * 4);
  unsigned short* efbf   = (unsigned short*)alloc((size_t)EH * 2);
  float*          rfused = (float*)alloc((size_t)RELN * 4);
  unsigned short* xbf    = (unsigned short*)alloc((size_t)BS * FLAT * 2);
  unsigned short* x2bf   = (unsigned short*)alloc((size_t)BS * H * 2);
  float*          stats  = (float*)alloc(2 * 515 * 4);

  // --- one-time converts / transposes (bf16 operand prep) ---
  k_cvt_bf16<<<(EH + 255) / 256, 256, 0, stream>>>(ent, entbf, EH);
  k_cvt_bf16<<<(2 * RELN + 255) / 256, 256, 0, stream>>>(rel, relbf, 2 * RELN);
  k_cvtT_bf16<<<256, 256, 0, stream>>>(S_w, swT, 256, 256);
  k_cvtT_bf16<<<256, 256, 0, stream>>>(L_w, lwT, 256, 256);
  for (int i = 0; i < 2; ++i) {
    k_cvtT_bf16<<<256, 256, 0, stream>>>(W   + (size_t)i * 65536, wT  + (size_t)i * 65536, 256, 256);
    k_cvtT_bf16<<<256, 256, 0, stream>>>(W_r + (size_t)i * 65536, wrT + (size_t)i * 65536, 256, 256);
    k_cvtT_bf16<<<256, 256, 0, stream>>>(nw  + (size_t)i * 65536, nwT + (size_t)i * 65536, 256, 256);
  }
  k_cvtT_bf16<<<(256 * FLAT + 255) / 256, 256, 0, stream>>>(fcw, fcT, FLAT, 256);
  k_zero_f32<<<5, 256, 0, stream>>>(stats, 2 * 515);

  // --- common / private projections (WMMA), e_fused init, rel_fused ---
  k_gemm_bt<0, true, true, true><<<2500, 256, 0, stream>>>(entbf, swT, S_b, common, commbf, NE, H, H);
  k_gemm_bt<0, true, true, true><<<2500, 256, 0, stream>>>(entbf, lwT, L_b, priv,   privbf, NE, H, H);
  k_copy_f32<<<(EH + 255) / 256, 256, 0, stream>>>(ent, efused, EH);
  k_relfused<<<(RELN + 255) / 256, 256, 0, stream>>>(rel, gate, rfused, RELN);

  // --- two composition layers, each over {common, private} ---
  for (int i = 0; i < 2; ++i) {
    const unsigned short* wTi  = wT  + (size_t)i * 65536;
    const unsigned short* wrTi = wrT + (size_t)i * 65536;
    const unsigned short* nwTi = nwT + (size_t)i * 65536;
    const float* a1 = a + (size_t)i * 768;
    const float* a2 = a1 + 256;
    const float* a3 = a1 + 512;

    k_gemm_bt<0, false, true, false><<<50, 256, 0, stream>>>(relbf + (size_t)i * RELN, wrTi,
                                                             nullptr, hrbuf, nullptr, 400, H, H);
    k_gemv2<<<50, 256, 0, stream>>>(hrbuf, a3, a3, ha3, dummy, 400);

    for (int pass = 0; pass < 2; ++pass) {
      const unsigned short* Abf = pass ? privbf : commbf;
      float* dst = pass ? cpb : ccb;
      k_gemm_bt<0, false, true, false><<<2500, 256, 0, stream>>>(Abf, wTi, nullptr, hbuf, nullptr, NE, H, H);
      k_gemv2<<<2500, 256, 0, stream>>>(hbuf, a1, a2, ha1, ha2, NE);
      k_edge_topk<<<2500, 256, 0, stream>>>(esrc, erel, ha1, ha2, ha3, hbuf, hrbuf, neighbf, NE);
      k_gemm_bt<1, false, true, false><<<2500, 256, 0, stream>>>(neighbf, nwTi, nullptr, dst, nullptr, NE, H, H);
    }
    k_stats<<<256, 256, 0, stream>>>(ccb, cpb, stats + (size_t)i * 515, NE);
    k_accum<<<(EH + 255) / 256, 256, 0, stream>>>(ccb, cpb, gate, i, efused, EH);
  }

  // --- decoder: conv -> fc(relu) -> score, plus corr finalize ---
  k_cvt_bf16<<<(EH + 255) / 256, 256, 0, stream>>>(efused, efbf, EH);
  k_conv<<<BS, 256, 0, stream>>>(h_id, r_id, efused, rfused, cw, cb, xbf);
  k_gemm_bt<2, true, false, true><<<64, 256, 0, stream>>>(xbf, fcT, fcb, nullptr, x2bf, BS, H, FLAT);

  float* out = (float*)d_out;
  k_gemm_bt<0, true, true, false><<<5000, 256, 0, stream>>>(x2bf, efbf, ebias, out, nullptr, BS, NE, H);
  k_finalize<<<1, 1, 0, stream>>>(stats, stats + 515, out + (size_t)BS * NE, NE);
}
